// GeoEncoderLayer_48533130445091
// MI455X (gfx1250) — matrix operand order repaired and checked by reading, not compiled, not run on hardware
//
#include <hip/hip_runtime.h>

// ---------------------------------------------------------------------------
// GeoEncoderLayer for MI455X (gfx1250, wave32, WMMA)
// B=8, N=2048, D=256, F=1024.  BF16 operands, FP32 WMMA accumulate.
// ---------------------------------------------------------------------------

typedef unsigned short u16;
typedef __attribute__((ext_vector_type(8)))  unsigned short v8u;
typedef __attribute__((ext_vector_type(16))) unsigned short v16u;
typedef __attribute__((ext_vector_type(16))) __bf16        v16bf;
typedef __attribute__((ext_vector_type(8)))  float         v8f;

constexpr int Bt = 8;
constexpr int Nn = 2048;
constexpr int Dd = 256;
constexpr int Ff = 1024;
constexpr int MN = Bt * Nn; // 16384 rows total

union U16V { v16u u; v16bf b; };

__device__ __forceinline__ u16 f2bf(float f) {
    unsigned int u = __float_as_uint(f);
    u += 0x7fffu + ((u >> 16) & 1u);   // round-to-nearest-even
    return (u16)(u >> 16);
}

__device__ __forceinline__ v8f zero8() {
    v8f z = {0.f, 0.f, 0.f, 0.f, 0.f, 0.f, 0.f, 0.f};
    return z;
}

// two 16B-aligned 8x u16 loads -> one 16-element fragment register group
__device__ __forceinline__ v16u ld16pair(const u16* p0, const u16* p1) {
    v8u a = *(const v8u*)p0;
    v8u b = *(const v8u*)p1;
    return __builtin_shufflevector(a, b, 0, 1, 2, 3, 4, 5, 6, 7,
                                         8, 9, 10, 11, 12, 13, 14, 15);
}

__device__ __forceinline__ v8f wmma_bf16(v16u a, v16u b, v8f c) {
    U16V ua; ua.u = a;
    U16V ub; ub.u = b;
    return __builtin_amdgcn_wmma_f32_16x16x32_bf16(
        false, ua.b, false, ub.b, (short)0, c, false, false);
}

// reductions across the 16-lane half-groups (matches C-frag N striping)
__device__ __forceinline__ float rmax16(float v) {
    #pragma unroll
    for (int m = 1; m < 16; m <<= 1) v = fmaxf(v, __shfl_xor(v, m, 16));
    return v;
}
__device__ __forceinline__ float rsum16(float v) {
    #pragma unroll
    for (int m = 1; m < 16; m <<= 1) v += __shfl_xor(v, m, 16);
    return v;
}

// ---------------------------------------------------------------------------
// weight transpose + fp32->bf16:  w[R][C] -> wT[C][R]
// ---------------------------------------------------------------------------
__global__ void transpose_bf16_kernel(const float* __restrict__ w,
                                      u16* __restrict__ wT, int R, int C) {
    int idx = blockIdx.x * blockDim.x + threadIdx.x;
    if (idx >= R * C) return;
    int r = idx / C, c = idx % C;
    wT[(size_t)c * R + r] = f2bf(w[idx]);
}

// ---------------------------------------------------------------------------
// LayerNorm over D=256, one block (256 threads) per row, bf16 output
// ---------------------------------------------------------------------------
__global__ __launch_bounds__(256)
void ln_bf16_kernel(const float* __restrict__ x, const float* __restrict__ gamma,
                    const float* __restrict__ beta, u16* __restrict__ out) {
    const int row = blockIdx.x;
    const int t = threadIdx.x;
    float v = x[(size_t)row * Dd + t];
    float s = v, q = v * v;
    #pragma unroll
    for (int m = 1; m < 32; m <<= 1) {
        s += __shfl_xor(s, m, 32);
        q += __shfl_xor(q, m, 32);
    }
    __shared__ float rs[8], rq[8], mu_s, rstd_s;
    const int wv = t >> 5, ln = t & 31;
    if (ln == 0) { rs[wv] = s; rq[wv] = q; }
    __syncthreads();
    if (t == 0) {
        float S = 0.f, Q = 0.f;
        #pragma unroll
        for (int i = 0; i < 8; i++) { S += rs[i]; Q += rq[i]; }
        float mu = S * (1.0f / Dd);
        float var = Q * (1.0f / Dd) - mu * mu;
        mu_s = mu;
        rstd_s = rsqrtf(var + 1e-5f);
    }
    __syncthreads();
    float o = (v - mu_s) * rstd_s * gamma[t] + beta[t];
    out[(size_t)row * Dd + t] = f2bf(o);
}

// ---------------------------------------------------------------------------
// Flash attention:  x1 = x + softmax(h h^T / 16) h
// grid (N/128, B), 256 threads (8 waves), each wave owns 16 query rows.
// ---------------------------------------------------------------------------
__global__ __launch_bounds__(256)
void attn_kernel(const u16* __restrict__ hb, const float* __restrict__ x,
                 float* __restrict__ x1) {
    __shared__ __align__(16) u16 vT[256][40];      // transposed V tile (pad 40)
    __shared__ __align__(16) u16 pbuf[8][16 * 32]; // per-wave P restaging

    const int b    = blockIdx.y;
    const int wv   = threadIdx.x >> 5;
    const int lane = threadIdx.x & 31;
    const int lh   = lane >> 4;   // lane half
    const int l16  = lane & 15;
    const int row_base = blockIdx.x * 128 + wv * 16;
    const u16* hB = hb + (size_t)b * Nn * Dd;

    v8f O[16];
    #pragma unroll
    for (int i = 0; i < 16; i++) O[i] = zero8();
    float mi[8], li[8];
    #pragma unroll
    for (int r = 0; r < 8; r++) { mi[r] = -__builtin_inff(); li[r] = 0.f; }

    const u16* qrow = hB + (size_t)(row_base + l16) * Dd;

    for (int ct = 0; ct < Nn / 32; ++ct) {
        const int col0 = ct * 32;
        // -- stage V tile transposed into LDS (all 256 threads) --
        {
            const int r  = threadIdx.x >> 3;         // 0..31 tile row
            const int c0 = (threadIdx.x & 7) * 32;   // 32-wide d segment
            const u16* src = hB + (size_t)(col0 + r) * Dd + c0;
            #pragma unroll
            for (int i0 = 0; i0 < 32; i0 += 8) {
                v8u d = *(const v8u*)(src + i0);
                #pragma unroll
                for (int i = 0; i < 8; i++) vT[c0 + i0 + i][r] = d[i];
            }
        }
        __syncthreads();

        // -- S = Q K^T over K=D=256 (8 bf16 WMMA chunks, two 16-col tiles) --
        v8f S0 = zero8(), S1 = zero8();
        const u16* c0p = hB + (size_t)(col0 + l16) * Dd;
        const u16* c1p = hB + (size_t)(col0 + 16 + l16) * Dd;
        #pragma unroll
        for (int kc = 0; kc < 8; kc++) {
            const int kb = kc * 32;
            v16u a  = ld16pair(qrow + kb + lh * 16, qrow + kb + lh * 16 + 8);
            v16u b0 = ld16pair(c0p + kb + lh * 16, c0p + kb + lh * 16 + 8);
            v16u b1 = ld16pair(c1p + kb + lh * 16, c1p + kb + lh * 16 + 8);
            S0 = wmma_bf16(a, b0, S0);
            S1 = wmma_bf16(a, b1, S1);
        }

        // -- online softmax (rows r+8*lh, cols striped over 16-lane group) --
        float p0[8], p1[8], alpha[8];
        #pragma unroll
        for (int r = 0; r < 8; r++) {
            float s0 = S0[r] * 0.0625f;
            float s1 = S1[r] * 0.0625f;
            float t = rmax16(fmaxf(s0, s1));
            float mn = fmaxf(mi[r], t);
            float al = __expf(mi[r] - mn);
            p0[r] = __expf(s0 - mn);
            p1[r] = __expf(s1 - mn);
            float srow = rsum16(p0[r] + p1[r]);
            li[r] = li[r] * al + srow;
            mi[r] = mn;
            alpha[r] = al;
        }
        #pragma unroll
        for (int dblk = 0; dblk < 16; dblk++) {
            #pragma unroll
            for (int r = 0; r < 8; r++) O[dblk][r] *= alpha[r];
        }

        // -- restage P through LDS: C-layout -> A-fragment layout --
        u16* pb = pbuf[wv];
        #pragma unroll
        for (int r = 0; r < 8; r++) {
            pb[(r + 8 * lh) * 32 + l16]      = f2bf(p0[r]);
            pb[(r + 8 * lh) * 32 + 16 + l16] = f2bf(p1[r]);
        }
        asm volatile("s_wait_dscnt 0" ::: "memory"); // wave-local LDS RAW
        v16u pA = ld16pair(pb + l16 * 32 + lh * 16,
                           pb + l16 * 32 + lh * 16 + 8);

        // -- O += P V  (K=32, 16 d-blocks of 16) --
        #pragma unroll
        for (int dblk = 0; dblk < 16; dblk++) {
            const u16* vp = &vT[dblk * 16 + l16][lh * 16];
            v16u bv = ld16pair(vp, vp + 8);
            O[dblk] = wmma_bf16(pA, bv, O[dblk]);
        }
        __syncthreads(); // protect vT before next tile overwrites it
    }

    // -- epilogue: x1 = x + O / l --
    #pragma unroll
    for (int r = 0; r < 8; r++) {
        const int row = row_base + r + 8 * lh;
        const float inv = 1.0f / li[r];
        const size_t base = ((size_t)b * Nn + row) * Dd;
        #pragma unroll
        for (int dblk = 0; dblk < 16; dblk++) {
            const int d = dblk * 16 + l16;
            x1[base + d] = x[base + d] + O[dblk][r] * inv;
        }
    }
}

// ---------------------------------------------------------------------------
// FFN GEMM 1: act = relu(h2 @ w1 + b1), bf16 out.  M=16384 K=256 N=1024
// grid (M/128, N/64); each wave -> 16x64 tile.
// ---------------------------------------------------------------------------
__global__ __launch_bounds__(256)
void ffn1_kernel(const u16* __restrict__ h2, const u16* __restrict__ w1T,
                 const float* __restrict__ b1, u16* __restrict__ act) {
    const int wv = threadIdx.x >> 5, lane = threadIdx.x & 31;
    const int lh = lane >> 4, l16 = lane & 15;
    const int row0 = blockIdx.x * 128 + wv * 16;
    const int col0 = blockIdx.y * 64;

    v8f acc[4] = {zero8(), zero8(), zero8(), zero8()};
    const u16* arow = h2 + (size_t)(row0 + l16) * Dd;
    #pragma unroll
    for (int kc = 0; kc < Dd / 32; kc++) {
        const int kb = kc * 32;
        v16u a = ld16pair(arow + kb + lh * 16, arow + kb + lh * 16 + 8);
        #pragma unroll
        for (int nt = 0; nt < 4; nt++) {
            const u16* bp = w1T + (size_t)(col0 + nt * 16 + l16) * Dd + kb + lh * 16;
            v16u bf = ld16pair(bp, bp + 8);
            acc[nt] = wmma_bf16(a, bf, acc[nt]);
        }
    }
    #pragma unroll
    for (int nt = 0; nt < 4; nt++) {
        const int col = col0 + nt * 16 + l16;
        const float bias = b1[col];
        #pragma unroll
        for (int r = 0; r < 8; r++) {
            const int row = row0 + r + 8 * lh;
            float v = fmaxf(acc[nt][r] + bias, 0.f);
            act[(size_t)row * Ff + col] = f2bf(v);
        }
    }
}

// ---------------------------------------------------------------------------
// FFN GEMM 2 + residual: out = x1 + act @ w2 + b2.  M=16384 K=1024 N=256
// ---------------------------------------------------------------------------
__global__ __launch_bounds__(256)
void ffn2_kernel(const u16* __restrict__ act, const u16* __restrict__ w2T,
                 const float* __restrict__ b2, const float* __restrict__ x1,
                 float* __restrict__ out) {
    const int wv = threadIdx.x >> 5, lane = threadIdx.x & 31;
    const int lh = lane >> 4, l16 = lane & 15;
    const int row0 = blockIdx.x * 128 + wv * 16;
    const int col0 = blockIdx.y * 64;

    v8f acc[4] = {zero8(), zero8(), zero8(), zero8()};
    const u16* arow = act + (size_t)(row0 + l16) * Ff;
    for (int kc = 0; kc < Ff / 32; kc++) {
        const int kb = kc * 32;
        v16u a = ld16pair(arow + kb + lh * 16, arow + kb + lh * 16 + 8);
        #pragma unroll
        for (int nt = 0; nt < 4; nt++) {
            const u16* bp = w2T + (size_t)(col0 + nt * 16 + l16) * Ff + kb + lh * 16;
            v16u bf = ld16pair(bp, bp + 8);
            acc[nt] = wmma_bf16(a, bf, acc[nt]);
        }
    }
    #pragma unroll
    for (int nt = 0; nt < 4; nt++) {
        const int col = col0 + nt * 16 + l16;
        const float bias = b2[col];
        #pragma unroll
        for (int r = 0; r < 8; r++) {
            const int row = row0 + r + 8 * lh;
            const size_t o = (size_t)row * Dd + col;
            out[o] = x1[o] + acc[nt][r] + bias;
        }
    }
}

// ---------------------------------------------------------------------------
extern "C" void kernel_launch(void* const* d_in, const int* in_sizes, int n_in,
                              void* d_out, int out_size, void* d_ws, size_t ws_size,
                              hipStream_t stream) {
    (void)in_sizes; (void)n_in; (void)out_size; (void)ws_size;
    const float* x   = (const float*)d_in[0];
    const float* w1  = (const float*)d_in[1];
    const float* b1  = (const float*)d_in[2];
    const float* w2  = (const float*)d_in[3];
    const float* b2  = (const float*)d_in[4];
    const float* g1  = (const float*)d_in[5];
    const float* be1 = (const float*)d_in[6];
    const float* g2  = (const float*)d_in[7];
    const float* be2 = (const float*)d_in[8];
    float* out = (float*)d_out;

    char* ws = (char*)d_ws;
    size_t off = 0;
    auto alloc = [&](size_t bytes) -> void* {
        void* p = ws + off;
        off += (bytes + 255) & ~(size_t)255;
        return p;
    };
    u16*   h_bf  = (u16*)  alloc((size_t)MN * Dd * 2);  //  8.4 MB
    float* x1    = (float*)alloc((size_t)MN * Dd * 4);  // 16.8 MB
    u16*   h2_bf = (u16*)  alloc((size_t)MN * Dd * 2);  //  8.4 MB
    u16*   actb  = (u16*)  alloc((size_t)MN * Ff * 2);  // 33.6 MB
    u16*   w1T   = (u16*)  alloc((size_t)Dd * Ff * 2);  //  0.5 MB
    u16*   w2T   = (u16*)  alloc((size_t)Ff * Dd * 2);  //  0.5 MB

    // weight transpose + bf16 convert (independent of LN/attn ordering)
    transpose_bf16_kernel<<<(Dd * Ff + 255) / 256, 256, 0, stream>>>(w1, w1T, Dd, Ff);
    transpose_bf16_kernel<<<(Ff * Dd + 255) / 256, 256, 0, stream>>>(w2, w2T, Ff, Dd);

    // sublayer 1
    ln_bf16_kernel<<<MN, 256, 0, stream>>>(x, g1, be1, h_bf);
    attn_kernel<<<dim3(Nn / 128, Bt), 256, 0, stream>>>(h_bf, x, x1);

    // sublayer 2
    ln_bf16_kernel<<<MN, 256, 0, stream>>>(x1, g2, be2, h2_bf);
    ffn1_kernel<<<dim3(MN / 128, Ff / 64), 256, 0, stream>>>(h2_bf, w1T, b1, actb);
    ffn2_kernel<<<dim3(MN / 128, Dd / 64), 256, 0, stream>>>(actb, w2T, b2, x1, out);
}
